// MultiheadAttention_89764816486741
// MI455X (gfx1250) — compile-verified, matrix operand order checked by reading, not run on hardware
//
#include <hip/hip_runtime.h>
#include <hip/hip_bf16.h>
#include <stdint.h>

// ---------------- problem constants ----------------
#define BB 4
#define SS 2048
#define DD 512
#define HH 8
#define HD 64
#define OO 512

typedef __attribute__((ext_vector_type(16))) _Float16 v16h;
typedef __attribute__((ext_vector_type(8)))  _Float16 v8h;
typedef __attribute__((ext_vector_type(8)))  float    v8f;

// ---------------- WMMA helpers ----------------
// D = A(16x32 f16) x B(32x16 f16) + C(16x16 f32)
__device__ __forceinline__ v8f wmma32(v16h a, v16h b, v8f c) {
    return __builtin_amdgcn_wmma_f32_16x16x32_f16(
        /*neg_a=*/false, a, /*neg_b=*/false, b,
        /*c_mod=*/(short)0, c, /*reuse_a=*/false, /*reuse_b=*/false);
}

// Unified fragment load (A from [m][k] row-major, B from [n][k] n-major).
// ISA 7.12.2: lanes 0-15 hold row/col = lane with K {0..7,16..23}; lanes 16-31
// hold row/col = lane-16 with K {8..15,24..31}. Two contiguous 16B chunks/lane.
__device__ __forceinline__ v16h load_frag(const _Float16* p, int stride, int lane) {
    const int row = lane & 15;
    const int kb  = (lane >> 4) << 3;          // 0 or 8
    const _Float16* r0 = p + row * stride + kb;
    v8h lo = *(const v8h*)(r0);
    v8h hi = *(const v8h*)(r0 + 16);
    return __builtin_shufflevector(lo, hi, 0, 1, 2, 3, 4, 5, 6, 7,
                                   8, 9, 10, 11, 12, 13, 14, 15);
}

// ---------------- gfx1250 async memory->LDS (ASYNCcnt path) ----------------
// Per ISA 15.18.3 / 08_async_tensor: each lane copies 16 bytes from its global address to its LDS offset.
__device__ __forceinline__ void async_copy_16B(uint32_t lds_off, const void* gptr) {
    asm volatile("global_load_async_to_lds_b128 %0, %1, off"
                 :: "v"(lds_off), "v"(gptr) : "memory");
}
__device__ __forceinline__ void wait_async0() {
    asm volatile("s_wait_asynccnt 0x0" ::: "memory");
}

// ---------------- weight convert + transpose ----------------
// src: [nh][512][C] f32 (C = 1<<cshift), dst: [nh][C][512] f16 (n-major for B frags)
__global__ void wtrans_kernel(const float* __restrict__ src,
                              _Float16* __restrict__ dst, int cshift) {
    int i = blockIdx.x * 256 + threadIdx.x;            // over nh*512*C == 262144
    int c = i & ((1 << cshift) - 1);
    int r = (i >> cshift) & 511;
    int hbase = (i >> (cshift + 9)) << (cshift + 9);   // h * 512 * C
    dst[(size_t)hbase + ((size_t)c << 9) + r] = (_Float16)src[i];
}

// ---------------- QKV projection ----------------
// Out at ((b*H+h)*S*HD) + s*stride_s + e*stride_e (f16)
//   q/k: (HD,1) -> [B,H,S,HD];  v: (1,S) -> [B,H,HD,S] (pre-transposed for PV B-frags)
__global__ void __launch_bounds__(128)
proj_qkv_kernel(const float* __restrict__ X, const _Float16* __restrict__ Wt,
                const float* __restrict__ bias, _Float16* __restrict__ Out,
                float oscale, int stride_s, int stride_e) {
    __shared__ __align__(16) _Float16 xs[64 * 40];     // [m][k] pad 40
    __shared__ __align__(16) _Float16 wsh[64 * 40];    // [col][k] pad 40
    const int tid = threadIdx.x, lane = tid & 31, wave = tid >> 5;
    const int row0 = blockIdx.x * 64;
    const int col0 = blockIdx.y * 64;
    const int h = col0 >> 6;

    v8f acc[4] = {};
    for (int d0 = 0; d0 < DD; d0 += 32) {
        __syncthreads();
        for (int i = tid; i < 64 * 4; i += 128) {      // X tile 64x32 f32->f16
            int r = i >> 2, c8 = (i & 3) * 8;
            const float* xp = X + (size_t)(row0 + r) * DD + d0 + c8;
            v8h h8;
#pragma unroll
            for (int j = 0; j < 8; ++j) h8[j] = (_Float16)xp[j];
            *(v8h*)(xs + r * 40 + c8) = h8;
        }
        const _Float16* wp = Wt + ((size_t)col0 << 9) + d0;
        for (int i = tid; i < 64 * 4; i += 128) {      // W tile [col][32]
            int lc = i >> 2, c8 = (i & 3) * 8;
            *(v8h*)(wsh + lc * 40 + c8) = *(const v8h*)(wp + ((size_t)lc << 9) + c8);
        }
        __syncthreads();

        v16h a = load_frag(xs + wave * 16 * 40, 40, lane);
#pragma unroll
        for (int nt = 0; nt < 4; ++nt)
            acc[nt] = wmma32(a, load_frag(wsh + nt * 16 * 40, 40, lane), acc[nt]);
    }
    const int m0 = (lane >> 4) << 3;
    const int nn = lane & 15;
    const size_t obase = (size_t)(blockIdx.y) * SS * HD;
#pragma unroll
    for (int nt = 0; nt < 4; ++nt) {
        int e = nt * 16 + nn;
        float bv = bias[h * HD + e];
#pragma unroll
        for (int r = 0; r < 8; ++r) {
            int grow = row0 + wave * 16 + m0 + r;
            int b = grow >> 11, s = grow & (SS - 1);
            float val = (acc[nt][r] + bv) * oscale;
            Out[(size_t)b * HH * SS * HD + obase +
                (size_t)s * stride_s + (size_t)e * stride_e] = (_Float16)val;
        }
    }
}

// ---------------- flash attention (async double-buffered tiles) ----------------
__global__ void __launch_bounds__(128)
flash_attn_kernel(const _Float16* __restrict__ qh, const _Float16* __restrict__ kh,
                  const _Float16* __restrict__ vht, const unsigned char* __restrict__ mask,
                  float* __restrict__ ctx) {
    __shared__ __align__(16) _Float16 ktb[2][64 * 72]; // [t][e] (n-major for score B)
    __shared__ __align__(16) _Float16 vtb[2][64 * 72]; // [e][t] (n-major for PV B)
    __shared__ __align__(16) _Float16 ps[4 * 16 * 72]; // per-wave P tile [m][t]

    const int tid = threadIdx.x, lane = tid & 31, wave = tid >> 5;
    const int s0 = blockIdx.x * 64;
    const int bh = blockIdx.y;
    const int b = bh >> 3;                              // H == 8
    const size_t base = (size_t)bh * SS * HD;
    const _Float16* kbase = kh + base;
    const _Float16* vbase = vht + base;

    // Q fragments straight from global (per-lane global_load_b128 x4)
    const int qrow = wave * 16;
    const _Float16* qp = qh + base + (size_t)(s0 + qrow + (lane & 15)) * HD + ((lane >> 4) << 3);
    v8h q00 = *(const v8h*)(qp);
    v8h q01 = *(const v8h*)(qp + 16);
    v8h q10 = *(const v8h*)(qp + 32);
    v8h q11 = *(const v8h*)(qp + 48);
    const v16h qa0 = __builtin_shufflevector(q00, q01, 0,1,2,3,4,5,6,7,8,9,10,11,12,13,14,15);
    const v16h qa1 = __builtin_shufflevector(q10, q11, 0,1,2,3,4,5,6,7,8,9,10,11,12,13,14,15);

    v8f acc[4] = {};
    float m_run[8], l_run[8];
#pragma unroll
    for (int r = 0; r < 8; ++r) { m_run[r] = -1e30f; l_run[r] = 0.f; }

    const int m0 = (lane >> 4) << 3;
    const int nn = lane & 15;
    const int myrow0 = s0 + qrow;
    _Float16* pw = ps + wave * 16 * 72;

    // async prefetch of a (K,V) chunk into buffer `buf`
    auto prefetch = [&](int buf, int t0) {
        const _Float16* kg = kbase + (size_t)t0 * HD;   // [t][e], row stride HD
        const _Float16* vg = vbase + t0;                // [e][t], row stride SS
        for (int i = tid; i < 512; i += 128) {
            int t = i >> 3, c8 = (i & 7) * 8;
            async_copy_16B((uint32_t)(uintptr_t)(&ktb[buf][t * 72 + c8]),
                           kg + t * HD + c8);
        }
        for (int i = tid; i < 512; i += 128) {
            int e = i >> 3, c8 = (i & 7) * 8;
            async_copy_16B((uint32_t)(uintptr_t)(&vtb[buf][e * 72 + c8]),
                           vg + ((size_t)e << 11) + c8);
        }
    };

    prefetch(0, 0);                                     // chunk 0 -> buffer 0

    for (int t0 = 0; t0 <= s0; t0 += 64) {              // causal chunks
        const int cur = (t0 >> 6) & 1;
        const _Float16* kt = ktb[cur];
        const _Float16* vt = vtb[cur];

        wait_async0();          // my async loads for buf[cur] complete
        __syncthreads();        // everyone's complete; prior readers of buf[cur^1] done
        if (t0 + 64 <= s0)
            prefetch(cur ^ 1, t0 + 64);                 // overlaps compute below

        // padding mask bytes for this chunk (tiny, L2-resident)
        bool kval[4];
#pragma unroll
        for (int nt = 0; nt < 4; ++nt)
            kval[nt] = mask[(size_t)b * SS + t0 + nt * 16 + nn] != 0;

        // scores: q(16x64) x K^T -> 4 C tiles (8 wmma)
        v8f sc[4];
#pragma unroll
        for (int nt = 0; nt < 4; ++nt) {
            v8f z = {};
            z = wmma32(qa0, load_frag(kt + nt * 16 * 72, 72, lane), z);
            z = wmma32(qa1, load_frag(kt + nt * 16 * 72 + 32, 72, lane), z);
            sc[nt] = z;
        }

        // causal + padding mask, chunk row max
        float nm[8];
#pragma unroll
        for (int r = 0; r < 8; ++r) nm[r] = m_run[r];
#pragma unroll
        for (int nt = 0; nt < 4; ++nt) {
            int n_abs = t0 + nt * 16 + nn;
#pragma unroll
            for (int r = 0; r < 8; ++r) {
                int m_abs = myrow0 + m0 + r;
                float v = sc[nt][r];
                if (!kval[nt] || (n_abs > m_abs)) v = -1e30f;
                sc[nt][r] = v;
                nm[r] = fmaxf(nm[r], v);
            }
        }
#pragma unroll
        for (int r = 0; r < 8; ++r) {                   // 16-lane row-group max
            float v = nm[r];
            v = fmaxf(v, __shfl_xor(v, 1, 32));
            v = fmaxf(v, __shfl_xor(v, 2, 32));
            v = fmaxf(v, __shfl_xor(v, 4, 32));
            v = fmaxf(v, __shfl_xor(v, 8, 32));
            nm[r] = v;
        }
        // P = exp(s-m) -> per-wave LDS [m][t] (A-frag friendly)
        float rs[8];
#pragma unroll
        for (int r = 0; r < 8; ++r) rs[r] = 0.f;
#pragma unroll
        for (int nt = 0; nt < 4; ++nt) {
#pragma unroll
            for (int r = 0; r < 8; ++r) {
                float p = __expf(sc[nt][r] - nm[r]);
                rs[r] += p;
                pw[(m0 + r) * 72 + nt * 16 + nn] = (_Float16)p;
            }
        }
        // row sums + online rescale
#pragma unroll
        for (int r = 0; r < 8; ++r) {
            float v = rs[r];
            v += __shfl_xor(v, 1, 32);
            v += __shfl_xor(v, 2, 32);
            v += __shfl_xor(v, 4, 32);
            v += __shfl_xor(v, 8, 32);
            float so = __expf(m_run[r] - nm[r]);
            l_run[r] = l_run[r] * so + v;
            m_run[r] = nm[r];
#pragma unroll
            for (int nt = 0; nt < 4; ++nt) acc[nt][r] *= so;
        }
        asm volatile("s_wait_dscnt 0x0" ::: "memory");  // wave-private P RAW fence
        v16h pa0 = load_frag(pw, 72, lane);
        v16h pa1 = load_frag(pw + 32, 72, lane);
#pragma unroll
        for (int nt = 0; nt < 4; ++nt) {                // PV (8 wmma)
            acc[nt] = wmma32(pa0, load_frag(vt + nt * 16 * 72, 72, lane), acc[nt]);
            acc[nt] = wmma32(pa1, load_frag(vt + nt * 16 * 72 + 32, 72, lane), acc[nt]);
        }
    }

    // normalize + store ctx f32 [B,S,H*HD]
#pragma unroll
    for (int r = 0; r < 8; ++r) {
        int s = myrow0 + m0 + r;
        float inv = 1.0f / l_run[r];
#pragma unroll
        for (int nt = 0; nt < 4; ++nt) {
            int e = nt * 16 + nn;
            ctx[((size_t)b * SS + s) * (HH * HD) + (bh & 7) * HD + e] = acc[nt][r] * inv;
        }
    }
}

// ---------------- output projection ----------------
__global__ void __launch_bounds__(128)
out_proj_kernel(const float* __restrict__ Xc, const _Float16* __restrict__ Wot,
                const float* __restrict__ bo, float* __restrict__ out) {
    __shared__ __align__(16) _Float16 xs[64 * 40];
    __shared__ __align__(16) _Float16 wsh[64 * 40];
    const int tid = threadIdx.x, lane = tid & 31, wave = tid >> 5;
    const int row0 = blockIdx.x * 64, col0 = blockIdx.y * 64;

    v8f acc[4] = {};
    for (int d0 = 0; d0 < HH * HD; d0 += 32) {
        __syncthreads();
        for (int i = tid; i < 64 * 4; i += 128) {
            int r = i >> 2, c8 = (i & 3) * 8;
            const float* xp = Xc + (size_t)(row0 + r) * (HH * HD) + d0 + c8;
            v8h h8;
#pragma unroll
            for (int j = 0; j < 8; ++j) h8[j] = (_Float16)xp[j];
            *(v8h*)(xs + r * 40 + c8) = h8;
        }
        const _Float16* wp = Wot + ((size_t)col0 << 9) + d0;
        for (int i = tid; i < 64 * 4; i += 128) {
            int lc = i >> 2, c8 = (i & 3) * 8;
            *(v8h*)(wsh + lc * 40 + c8) = *(const v8h*)(wp + ((size_t)lc << 9) + c8);
        }
        __syncthreads();
        v16h a = load_frag(xs + wave * 16 * 40, 40, lane);
#pragma unroll
        for (int nt = 0; nt < 4; ++nt)
            acc[nt] = wmma32(a, load_frag(wsh + nt * 16 * 40, 40, lane), acc[nt]);
    }
    const int m0 = (lane >> 4) << 3, nn = lane & 15;
#pragma unroll
    for (int nt = 0; nt < 4; ++nt) {
        int c = col0 + nt * 16 + nn;
        float bv = bo[c];
#pragma unroll
        for (int r = 0; r < 8; ++r) {
            int grow = row0 + wave * 16 + m0 + r;
            out[(size_t)grow * OO + c] = acc[nt][r] + bv;
        }
    }
}

// ---------------- launcher ----------------
extern "C" void kernel_launch(void* const* d_in, const int* in_sizes, int n_in,
                              void* d_out, int out_size, void* d_ws, size_t ws_size,
                              hipStream_t stream) {
    (void)in_sizes; (void)n_in; (void)out_size; (void)ws_size;
    const float* Q  = (const float*)d_in[0];
    const float* K  = (const float*)d_in[1];
    const float* V  = (const float*)d_in[2];
    const unsigned char* mask = (const unsigned char*)d_in[3];
    const float* Wq = (const float*)d_in[4];
    const float* bq = (const float*)d_in[5];
    const float* Wk = (const float*)d_in[6];
    const float* bk = (const float*)d_in[7];
    const float* Wv = (const float*)d_in[8];
    const float* bv = (const float*)d_in[9];
    const float* Wo = (const float*)d_in[10];
    const float* bo = (const float*)d_in[11];

    char* ws = (char*)d_ws;
    const size_t nQKVh = (size_t)BB * HH * SS * HD;          // 4,194,304
    const size_t nW    = (size_t)HH * DD * HD;               // 262,144
    _Float16* qh    = (_Float16*)(ws);
    _Float16* kh    = (_Float16*)(ws + nQKVh * 2);
    _Float16* vht   = (_Float16*)(ws + nQKVh * 4);
    float*    ctx   = (float*)   (ws + nQKVh * 6);
    _Float16* wq16t = (_Float16*)(ws + nQKVh * 6 + nQKVh * 2 * sizeof(float));
    _Float16* wk16t = wq16t + nW;
    _Float16* wv16t = wk16t + nW;
    _Float16* wo16t = wv16t + nW;

    // 1) weight convert + transpose to [col][d] f16
    wtrans_kernel<<<1024, 256, 0, stream>>>(Wq, wq16t, 6);
    wtrans_kernel<<<1024, 256, 0, stream>>>(Wk, wk16t, 6);
    wtrans_kernel<<<1024, 256, 0, stream>>>(Wv, wv16t, 6);
    wtrans_kernel<<<1024, 256, 0, stream>>>(Wo, wo16t, 9);

    // 2) projections (q pre-scaled by 1/sqrt(HD)); v written transposed [B,H,HD,S]
    dim3 gproj(BB * SS / 64, (HH * HD) / 64);
    proj_qkv_kernel<<<gproj, 128, 0, stream>>>(Q, wq16t, bq, qh, 0.125f, HD, 1);
    proj_qkv_kernel<<<gproj, 128, 0, stream>>>(K, wk16t, bk, kh, 1.0f, HD, 1);
    proj_qkv_kernel<<<gproj, 128, 0, stream>>>(V, wv16t, bv, vht, 1.0f, 1, SS);

    // 3) flash attention
    flash_attn_kernel<<<dim3(SS / 64, BB * HH), 128, 0, stream>>>(qh, kh, vht, mask, ctx);

    // 4) output projection
    out_proj_kernel<<<dim3(BB * SS / 64, OO / 64), 128, 0, stream>>>(ctx, wo16t, bo, (float*)d_out);
}
